// CausalSelfAttention_13091060318984
// MI455X (gfx1250) — compile-verified, matrix operand order checked by reading, not run on hardware
//
#include <hip/hip_runtime.h>
#include <hip/hip_bf16.h>
#include <math.h>

// ---------------------------------------------------------------------------
// CausalSelfAttention for MI455X (gfx1250, wave32, WMMA + TDM).
// fp32 I/O; bf16 WMMA with f32 accumulation; bf16 intermediates staged into
// LDS with tensor_load_to_lds (Tensor Data Mover, TENSORcnt) where the tile
// is a plain strided-row copy.
// ---------------------------------------------------------------------------

typedef __attribute__((ext_vector_type(16))) __bf16 v16bf;
typedef __attribute__((ext_vector_type(8)))  __bf16 v8bf;
typedef __attribute__((ext_vector_type(4)))  __bf16 v4bf;
typedef __attribute__((ext_vector_type(8)))  float  v8f;
typedef __attribute__((ext_vector_type(4)))  unsigned int v4u;
typedef __attribute__((ext_vector_type(8)))  int    v8i;
typedef __attribute__((ext_vector_type(4)))  int    v4i;

static constexpr int Bc  = 2;
static constexpr int Tc  = 2048;
static constexpr int Cc  = 1024;
static constexpr int NH  = 16;
static constexpr int Dh  = 64;       // head dim
static constexpr int C3  = 3 * Cc;

// ---------------------------------------------------------------------------
// 16x32 bf16 A/B fragment (ISA 7.12.2): lane l holds row (l&15); half h=l>>4
// holds K = {8h..8h+7} in halves 0-7 and {8h+16..8h+23} in halves 8-15.
// ---------------------------------------------------------------------------
union Frag16 { v16bf v; v8bf h[2]; };
static __device__ inline v16bf load_frag(const __bf16* row, int kOff) {
    Frag16 f;
    f.h[0] = *(const v8bf*)(row + kOff);
    f.h[1] = *(const v8bf*)(row + kOff + 16);
    return f.v;
}

static __device__ inline v8f wmma_bf16(v16bf a, v16bf b, v8f c) {
    return __builtin_amdgcn_wmma_f32_16x16x32_bf16(
        false, a, false, b, (short)0, c, false, false);
}

static __device__ inline unsigned int pack2(__bf16 lo, __bf16 hi) {
    union { __bf16 b[2]; unsigned int u; } x;
    x.b[0] = lo; x.b[1] = hi;
    return x.u;
}

// Low 32 bits of a flat pointer into LDS == LDS byte offset (ISA 10.2:
// the LDS address is the flat address's low 32 bits).
static __device__ inline unsigned int lds_byte_off(const void* p) {
    return (unsigned int)(uintptr_t)p;
}

// ---------------------------------------------------------------------------
// TDM: async-load a 2D bf16 tile (tileW elems x tileH rows, row stride
// `strideElems`) from global into contiguous LDS at `ldsOff`.
// D# encoding per CDNA5 ISA (2D: groups 2/3 unused).
// This toolchain exposes the 6-arg builtin form:
//   (uint32x4 g0, int32x8 g1, int32x4, int32x4, int32x8, i32 cpol).
// Issue from one wave; completion via s_wait_tensorcnt.
// ---------------------------------------------------------------------------
static __device__ inline void tdm_load_2d_bf16(unsigned int ldsOff,
                                               const void* gsrc,
                                               unsigned int tileW,
                                               unsigned int tileH,
                                               unsigned long long strideElems) {
    unsigned long long ga = (unsigned long long)(uintptr_t)gsrc;
    const unsigned int dim0 = 1u << 20, dim1 = 1u << 20;  // oversized: no OOB clip
    v4u g0;
    g0[0] = 1u;                                            // count=1 (valid), user mode
    g0[1] = ldsOff;                                        // lds_addr [63:32]
    g0[2] = (unsigned int)(ga & 0xFFFFFFFFu);              // global_addr lo
    g0[3] = (unsigned int)((ga >> 32) & 0x01FFFFFFu)       // global_addr [56:32]
          | (2u << 30);                                    // type=2 ("image")
    v8i g1;
    g1[0] = (int)(1u << 16);                               // wg_mask=0 | data_size=1 (2B)
    g1[1] = (int)((dim0 & 0xFFFFu) << 16);                 // barrier_addr=0 | dim0[15:0]
    g1[2] = (int)((dim0 >> 16) | ((dim1 & 0xFFFFu) << 16));// dim0[31:16] | dim1[15:0]
    g1[3] = (int)((dim1 >> 16) | (tileW << 16));           // dim1[31:16] | tile_dim0
    g1[4] = (int)(tileH & 0xFFFFu);                        // tile_dim1 | tile_dim2=0
    g1[5] = (int)(unsigned int)(strideElems & 0xFFFFFFFFull);      // dim0_stride lo
    g1[6] = (int)(unsigned int)((strideElems >> 32) & 0xFFFFull);  // dim0_stride hi | dim1_stride=0
    g1[7] = 0;
    v4i z4; z4[0] = 0; z4[1] = 0; z4[2] = 0; z4[3] = 0;
    v8i z8; z8[0] = 0; z8[1] = 0; z8[2] = 0; z8[3] = 0;
    z8[4] = 0; z8[5] = 0; z8[6] = 0; z8[7] = 0;
    __builtin_amdgcn_tensor_load_to_lds(g0, g1, z4, z4, z8, 0);
}

// ---------------------------------------------------------------------------
// GEMM: C[M,N] = A[M,K] * Bw[K,N].  A: fp32 (convert path) or bf16 (TDM path).
// Bw: fp32, transposed into LDS [n][k] with packed bf16x2 stores.
// Out: fp32 or bf16.  Block tile 128x128, 8 waves = 4(M) x 2(N), wave 32x64.
// ---------------------------------------------------------------------------
template <typename TA, typename TO>
__global__ __launch_bounds__(256) void gemm_wmma(const TA* __restrict__ A,
                                                 const float* __restrict__ Bw,
                                                 TO* __restrict__ C,
                                                 int M, int N, int K) {
    __shared__ __bf16 As[128][32];   // [m][k]
    __shared__ __bf16 Bs[128][32];   // [n][k]
    constexpr bool A_IS_BF16 = (sizeof(TA) == 2);

    const int tid  = threadIdx.x;
    const int wave = tid >> 5;
    const int lane = tid & 31;
    const int r    = lane & 15;
    const int hh   = lane >> 4;
    const int kOff = hh * 8;
    const int wm   = (wave & 3) * 32;
    const int wn   = (wave >> 2) * 64;
    const int blockM = blockIdx.y * 128;
    const int blockN = blockIdx.x * 128;

    v8f acc[2][4];
#pragma unroll
    for (int mt = 0; mt < 2; ++mt)
#pragma unroll
        for (int nt = 0; nt < 4; ++nt)
#pragma unroll
            for (int e = 0; e < 8; ++e) acc[mt][nt][e] = 0.0f;

    const int arow = tid >> 1, acol = (tid & 1) * 16;        // fp32 A loader
    const int bk0  = (tid >> 4) * 2, bn0 = (tid & 15) * 8;   // B loader (k-pairs)

    for (int k0 = 0; k0 < K; k0 += 32) {
        __syncthreads();
        // ---- stage A tile -> As[m][k] ----
        if constexpr (A_IS_BF16) {
            if (wave == 0)
                tdm_load_2d_bf16(lds_byte_off(&As[0][0]),
                                 A + (size_t)blockM * K + k0,
                                 32u, 128u, (unsigned long long)K);
        } else {
            const float* src = (const float*)A + (size_t)(blockM + arow) * K + k0 + acol;
            if (k0 + 32 < K) __builtin_prefetch(src + 32, 0, 0);
#pragma unroll
            for (int i = 0; i < 16; i += 4) {
                float4 v = *(const float4*)(src + i);
                As[arow][acol + i + 0] = (__bf16)v.x;
                As[arow][acol + i + 1] = (__bf16)v.y;
                As[arow][acol + i + 2] = (__bf16)v.z;
                As[arow][acol + i + 3] = (__bf16)v.w;
            }
        }
        // ---- stage B tile -> Bs[n][k] (transpose, packed k-pair stores) ----
        {
            const float* s0 = Bw + (size_t)(k0 + bk0) * N + blockN + bn0;
            const float* s1 = s0 + N;
            float4 x0 = *(const float4*)(s0);
            float4 x1 = *(const float4*)(s0 + 4);
            float4 y0 = *(const float4*)(s1);
            float4 y1 = *(const float4*)(s1 + 4);
            *(unsigned int*)&Bs[bn0 + 0][bk0] = pack2((__bf16)x0.x, (__bf16)y0.x);
            *(unsigned int*)&Bs[bn0 + 1][bk0] = pack2((__bf16)x0.y, (__bf16)y0.y);
            *(unsigned int*)&Bs[bn0 + 2][bk0] = pack2((__bf16)x0.z, (__bf16)y0.z);
            *(unsigned int*)&Bs[bn0 + 3][bk0] = pack2((__bf16)x0.w, (__bf16)y0.w);
            *(unsigned int*)&Bs[bn0 + 4][bk0] = pack2((__bf16)x1.x, (__bf16)y1.x);
            *(unsigned int*)&Bs[bn0 + 5][bk0] = pack2((__bf16)x1.y, (__bf16)y1.y);
            *(unsigned int*)&Bs[bn0 + 6][bk0] = pack2((__bf16)x1.z, (__bf16)y1.z);
            *(unsigned int*)&Bs[bn0 + 7][bk0] = pack2((__bf16)x1.w, (__bf16)y1.w);
        }
        if constexpr (A_IS_BF16) {
            if (wave == 0) __builtin_amdgcn_s_wait_tensorcnt(0);
        }
        __syncthreads();

        // ---- fragments + 8 WMMAs ----
        v16bf aF[2], bF[4];
#pragma unroll
        for (int mt = 0; mt < 2; ++mt)
            aF[mt] = load_frag(&As[wm + mt * 16 + r][0], kOff);
#pragma unroll
        for (int nt = 0; nt < 4; ++nt)
            bF[nt] = load_frag(&Bs[wn + nt * 16 + r][0], kOff);
#pragma unroll
        for (int mt = 0; mt < 2; ++mt)
#pragma unroll
            for (int nt = 0; nt < 4; ++nt)
                acc[mt][nt] = wmma_bf16(aF[mt], bF[nt], acc[mt][nt]);
    }

    // C/D layout: half hh -> rows {8hh..8hh+7} in VGPR 0..7, col = r.
#pragma unroll
    for (int mt = 0; mt < 2; ++mt) {
        const int rowBase = blockM + wm + mt * 16 + hh * 8;
#pragma unroll
        for (int nt = 0; nt < 4; ++nt) {
            const int col = blockN + wn + nt * 16 + r;
#pragma unroll
            for (int i = 0; i < 8; ++i)
                C[(size_t)(rowBase + i) * N + col] = (TO)acc[mt][nt][i];
        }
    }
}

// ---------------------------------------------------------------------------
// Flash-style causal attention over bf16 qkv. Grid: (T/128, NH, B).
// 8 waves/block, wave owns 16 query rows, keys stream in steps of 32.
// Q and K tiles staged by TDM (contiguous-row copies); V^T staged with
// cooperative packed transpose.
// ---------------------------------------------------------------------------
__global__ __launch_bounds__(256) void attn_kernel(const __bf16* __restrict__ qkv,
                                                   __bf16* __restrict__ y) {
    __shared__ __bf16 Qs[128][Dh];       // 16 KB  query tile
    __shared__ __bf16 Ks[32][Dh];        //  4 KB  [key][dim]  == B frag for Q.K^T
    __shared__ __bf16 Vt[Dh][32];        //  4 KB  [dim][key]  == B frag for P.V
    __shared__ __bf16 Ps[8][16][32];     //  8 KB  per-wave P (C-layout -> A-layout)

    const int tid  = threadIdx.x;
    const int wave = tid >> 5;
    const int lane = tid & 31;
    const int r    = lane & 15;
    const int hh   = lane >> 4;
    const int kOff = hh * 8;

    const int b = blockIdx.z, h = blockIdx.y;
    const int qBlock = blockIdx.x * 128;

    const __bf16* qBase = qkv + (size_t)b * Tc * C3 + h * Dh;
    const __bf16* kBase = qBase + Cc;
    const __bf16* vBase = qBase + 2 * Cc;

    // Stage the block's 128x64 Q tile via TDM.
    if (wave == 0) {
        tdm_load_2d_bf16(lds_byte_off(&Qs[0][0]),
                         qBase + (size_t)qBlock * C3, (unsigned)Dh, 128u,
                         (unsigned long long)C3);
        __builtin_amdgcn_s_wait_tensorcnt(0);
    }
    __syncthreads();

    const int qRow0 = wave * 16;
    v16bf aQ0 = load_frag(&Qs[qRow0 + r][0], kOff);
    v16bf aQ1 = load_frag(&Qs[qRow0 + r][0], 32 + kOff);

    v8f o[4];
#pragma unroll
    for (int nt = 0; nt < 4; ++nt)
#pragma unroll
        for (int e = 0; e < 8; ++e) o[nt][e] = 0.0f;
    float ms[8], ls[8];
#pragma unroll
    for (int i = 0; i < 8; ++i) { ms[i] = -INFINITY; ls[i] = 0.0f; }

    const int   qG0    = qBlock + qRow0;
    const int   myRow0 = qG0 + hh * 8;
    const int   kEnd   = qBlock + 128;
    const float scale  = 0.125f;          // 1/sqrt(64)

    const int vk0 = (tid >> 4) * 2;       // V^T loader: key-pair base
    const int vd0 = (tid & 15) * 4;       // 4 dims per thread

    for (int kb = 0; kb < kEnd; kb += 32) {
        __syncthreads();
        // K tile (32x64 rows) via TDM -> Ks[key][dim].
        if (wave == 0)
            tdm_load_2d_bf16(lds_byte_off(&Ks[0][0]),
                             kBase + (size_t)kb * C3, (unsigned)Dh, 32u,
                             (unsigned long long)C3);
        // V^T cooperative transpose (bf16 reads, packed key-pair b32 stores).
        {
            const __bf16* s0 = vBase + (size_t)(kb + vk0) * C3 + vd0;
            const __bf16* s1 = s0 + C3;
            v4bf a = *(const v4bf*)s0;
            v4bf c = *(const v4bf*)s1;
#pragma unroll
            for (int j = 0; j < 4; ++j)
                *(unsigned int*)&Vt[vd0 + j][vk0] = pack2(a[j], c[j]);
        }
        if (wave == 0) __builtin_amdgcn_s_wait_tensorcnt(0);
        __syncthreads();

        if (kb <= qG0 + 15) {  // step intersects this wave's causal range
            v8f S[2];
#pragma unroll
            for (int sub = 0; sub < 2; ++sub) {
                v8f s;
#pragma unroll
                for (int e = 0; e < 8; ++e) s[e] = 0.0f;
                v16bf bK0 = load_frag(&Ks[sub * 16 + r][0], kOff);
                v16bf bK1 = load_frag(&Ks[sub * 16 + r][0], 32 + kOff);
                s = wmma_bf16(aQ0, bK0, s);
                s = wmma_bf16(aQ1, bK1, s);
                S[sub] = s;
            }
            float rowmax[8];
#pragma unroll
            for (int i = 0; i < 8; ++i) rowmax[i] = -INFINITY;
#pragma unroll
            for (int sub = 0; sub < 2; ++sub) {
                const int key = kb + sub * 16 + r;
#pragma unroll
                for (int i = 0; i < 8; ++i) {
                    float sv = S[sub][i] * scale;
                    sv = (key <= myRow0 + i) ? sv : -INFINITY;
                    S[sub][i] = sv;
                    rowmax[i] = fmaxf(rowmax[i], sv);
                }
            }
#pragma unroll
            for (int i = 0; i < 8; ++i) {
                float v = rowmax[i];
                v = fmaxf(v, __shfl_xor(v, 1, 32));
                v = fmaxf(v, __shfl_xor(v, 2, 32));
                v = fmaxf(v, __shfl_xor(v, 4, 32));
                v = fmaxf(v, __shfl_xor(v, 8, 32));
                rowmax[i] = v;
            }
            float corr[8], rsum[8];
#pragma unroll
            for (int i = 0; i < 8; ++i) {
                float mnew = fmaxf(ms[i], rowmax[i]);
                float mc   = fmaxf(mnew, -1e30f);
                corr[i] = __expf(ms[i] - mc);
                ms[i]   = mnew;
                rsum[i] = 0.0f;
            }
#pragma unroll
            for (int sub = 0; sub < 2; ++sub)
#pragma unroll
                for (int i = 0; i < 8; ++i) {
                    float mc = fmaxf(ms[i], -1e30f);
                    float p  = __expf(S[sub][i] - mc);
                    rsum[i] += p;
                    Ps[wave][hh * 8 + i][sub * 16 + r] = (__bf16)p;
                }
#pragma unroll
            for (int i = 0; i < 8; ++i) {
                float v = rsum[i];
                v += __shfl_xor(v, 1, 32);
                v += __shfl_xor(v, 2, 32);
                v += __shfl_xor(v, 4, 32);
                v += __shfl_xor(v, 8, 32);
                ls[i] = ls[i] * corr[i] + v;
            }
#pragma unroll
            for (int nt = 0; nt < 4; ++nt)
#pragma unroll
                for (int i = 0; i < 8; ++i) o[nt][i] *= corr[i];

            v16bf aP = load_frag(&Ps[wave][r][0], kOff);
#pragma unroll
            for (int nt = 0; nt < 4; ++nt) {
                v16bf bV = load_frag(&Vt[nt * 16 + r][0], kOff);
                o[nt] = wmma_bf16(aP, bV, o[nt]);
            }
        }
    }

    // Normalize and store y[b, t, h*Dh + d] (bf16).
#pragma unroll
    for (int nt = 0; nt < 4; ++nt) {
        const int col = h * Dh + nt * 16 + r;
#pragma unroll
        for (int i = 0; i < 8; ++i) {
            const int row = qG0 + hh * 8 + i;
            y[(size_t)b * Tc * Cc + (size_t)row * Cc + col] = (__bf16)(o[nt][i] / ls[i]);
        }
    }
}

// ---------------------------------------------------------------------------
extern "C" void kernel_launch(void* const* d_in, const int* in_sizes, int n_in,
                              void* d_out, int out_size, void* d_ws, size_t ws_size,
                              hipStream_t stream) {
    (void)in_sizes; (void)n_in; (void)out_size;
    const float* x      = (const float*)d_in[0];   // [B, T, C]
    const float* w_qkv  = (const float*)d_in[1];   // [C, 3C]
    const float* w_proj = (const float*)d_in[2];   // [C, C]
    float*       out    = (float*)d_out;           // [B, T, C]

    const size_t qkvElems = (size_t)Bc * Tc * C3;  // 12.58M bf16 (25.2 MB)
    const size_t yElems   = (size_t)Bc * Tc * Cc;  //  4.19M bf16 ( 8.4 MB)
    if (ws_size < (qkvElems + yElems) * sizeof(__bf16)) return;
    __bf16* qkvBuf = (__bf16*)d_ws;
    __bf16* yBuf   = qkvBuf + qkvElems;

    const int M = Bc * Tc;  // 4096

    // 1) qkv = x @ w_qkv  (fp32 in, bf16 out): [4096,1024] x [1024,3072]
    dim3 g1(C3 / 128, M / 128);
    gemm_wmma<float, __bf16><<<g1, 256, 0, stream>>>(x, w_qkv, qkvBuf, M, C3, Cc);

    // 2) y = causal_attention(qkv)  (bf16 in, bf16 out)
    dim3 g2(Tc / 128, NH, Bc);
    attn_kernel<<<g2, 256, 0, stream>>>(qkvBuf, yBuf);

    // 3) out = y @ w_proj  (bf16 A via TDM, fp32 out): [4096,1024] x [1024,1024]
    dim3 g3(Cc / 128, M / 128);
    gemm_wmma<__bf16, float><<<g3, 256, 0, stream>>>(yBuf, w_proj, out, M, Cc, Cc);
}
